// SelfAttentionLayer_59811714564800
// MI455X (gfx1250) — matrix operand order repaired and checked by reading, not compiled, not run on hardware
//
#include <hip/hip_runtime.h>
#include <hip/hip_bf16.h>

typedef __bf16 bf16;
typedef __attribute__((ext_vector_type(16))) __bf16 bf16x16;
typedef __attribute__((ext_vector_type(8)))  __bf16 bf16x8;
typedef __attribute__((ext_vector_type(4)))  __bf16 bf16x4;
typedef __attribute__((ext_vector_type(8)))  float  f32x8;
typedef __attribute__((ext_vector_type(4)))  float  f32x4;
typedef __attribute__((ext_vector_type(4)))  unsigned int u32x4;
typedef __attribute__((ext_vector_type(8)))  int i32x8;
typedef __attribute__((ext_vector_type(4)))  int i32x4;

#if __has_builtin(__builtin_amdgcn_tensor_load_to_lds) && __has_builtin(__builtin_amdgcn_s_wait_tensorcnt)
#define USE_TDM 1
#else
#define USE_TDM 0
#warning "TDM builtins unavailable on this toolchain; using plain global loads"
#endif

union FragU { bf16x16 v; bf16x8 h[2]; };

// A-matrix fragment (16x32 bf16): lane<16 holds K {0..7,16..23}, lane>=16 holds {8..15,24..31}
// caller passes p = &row[halfsel*8]; we load +0 and +16 elements.
__device__ __forceinline__ bf16x16 ld_fragA(const bf16* p) {
  FragU u;
  u.h[0] = *(const bf16x8*)(p);
  u.h[1] = *(const bf16x8*)(p + 16);
  return u.v;
}

// B-matrix fragment (32x16 bf16): lane<16 holds K {0..7,16..23}, lane>=16 holds {8..15,24..31}
// caller passes p = &row[halfsel*8]; we load +0 and +16 elements.
__device__ __forceinline__ bf16x16 ld_fragB(const bf16* p) {
  FragU u;
  u.h[0] = *(const bf16x8*)(p);
  u.h[1] = *(const bf16x8*)(p + 16);
  return u.v;
}

__device__ __forceinline__ f32x8 wmma_bf16(bf16x16 a, bf16x16 b, f32x8 c) {
  return __builtin_amdgcn_wmma_f32_16x16x32_bf16(false, a, false, b, (short)0, c, false, false);
}

#if USE_TDM
// Tensor-DMA: load a 64x64 bf16 tile (row stride 1024 elements) into LDS at
// lds_off, padding 4 DWORDs (8 bf16) after every 32 DWORDs (64 bf16) so rows
// land with the same 72-element stride the WMMA fragment loaders expect.
// D# packing per CDNA5 ISA (2D: groups 2,3 = 0).
// This toolchain exposes the 6-arg builtin:
//   (u32x4 g0, i32x8 g1, i32x4 g2, i32x4 g3, i32x8 extra, i32 cpol)
__device__ __forceinline__ void tdm_load_tile64x64(unsigned lds_off, const void* gsrc,
                                                   unsigned tensor_rows) {
  unsigned long long ga = (unsigned long long)(size_t)gsrc;
  u32x4 g0;
  g0[0] = 1u;                                    // count=1 (valid), user mode
  g0[1] = lds_off;                               // lds_addr (bytes)
  g0[2] = (unsigned)ga;                          // global_addr[31:0]
  g0[3] = (unsigned)(ga >> 32) | 0x80000000u;    // global_addr[56:32] | type=2<<30
  i32x8 g1;
  // [17:16]=data_size(2B)=1, [20]=pad_enable, [24:22]=pad_interval(32dw)=4,
  // [31:25]=pad_amount(4dw)=3
  g1[0] = (int)((1u << 16) | (1u << 20) | (4u << 22) | (3u << 25));
  g1[1] = (int)(1024u << 16);                    // tensor_dim0[15:0]=1024 @bits79:48
  g1[2] = (int)(tensor_rows << 16);              // tensor_dim1[15:0]    @bits111:80
  g1[3] = (int)(64u << 16);                      // tile_dim0=64         @bits127:112
  g1[4] = 64;                                    // tile_dim1=64         @bits143:128
  g1[5] = 1024;                                  // tensor_dim0_stride=1024 @bits207:160
  g1[6] = 0;
  g1[7] = 0;
  i32x4 z4 = {0, 0, 0, 0};
  i32x8 z8 = {0, 0, 0, 0, 0, 0, 0, 0};
  __builtin_amdgcn_tensor_load_to_lds(g0, g1, z4, z4, z8, 0);
}
#endif

// ---------------------------------------------------------------------------
// C(M,N) = A(M,K) @ W(N,K)^T + bias(N), computed in bf16 WMMA, f32 accumulate.
// AT = float (convert on load) or bf16 (direct).  OT = bf16 or float.
// Block: 256 threads (8 waves), 128x128 tile, K-step 32.
// ---------------------------------------------------------------------------
template <typename AT, typename OT>
__global__ __launch_bounds__(256) void gemm_bt_bias(
    const AT* __restrict__ A, const float* __restrict__ W,
    const float* __restrict__ bias, OT* __restrict__ C,
    int M, int N, int K)
{
  constexpr int BM = 128, BN = 128, BK = 32, LDA = 40; // pad: 40 bf16 = 80 B rows (16B aligned)
  __shared__ bf16 As[BM * LDA];
  __shared__ bf16 Bs[BN * LDA];

  const int tid     = threadIdx.x;
  const int wave    = tid >> 5;
  const int lane    = tid & 31;
  const int lane16  = lane & 15;
  const int halfsel = lane >> 4;
  const int wm      = wave & 1;   // 2 wave-rows  (64 M each)
  const int wn      = wave >> 1;  // 4 wave-cols  (32 N each)
  const int bm0     = blockIdx.x * BM;
  const int bn0     = blockIdx.y * BN;

  f32x8 acc[4][2];
#pragma unroll
  for (int i = 0; i < 4; ++i)
#pragma unroll
    for (int j = 0; j < 2; ++j) acc[i][j] = {};

  for (int k0 = 0; k0 < K; k0 += BK) {
    __syncthreads();
    // ---- stage A tile (128 x 32) into LDS as bf16 ----
    if constexpr (sizeof(AT) == 4) {
      const int cg = (tid & 7) * 4;
#pragma unroll
      for (int p = 0; p < 4; ++p) {
        const int r = (tid >> 3) + p * 32;
        f32x4 d = *(const f32x4*)(A + (size_t)(bm0 + r) * K + k0 + cg);
        bf16x4 s;
        s[0] = (bf16)d[0]; s[1] = (bf16)d[1]; s[2] = (bf16)d[2]; s[3] = (bf16)d[3];
        *(bf16x4*)&As[r * LDA + cg] = s;
      }
    } else {
      const int cg = (tid & 3) * 8;
#pragma unroll
      for (int p = 0; p < 2; ++p) {
        const int r = (tid >> 2) + p * 64;
        bf16x8 d = *(const bf16x8*)(A + (size_t)(bm0 + r) * K + k0 + cg);
        *(bf16x8*)&As[r * LDA + cg] = d;
      }
    }
    // ---- stage W tile (128 rows of N, 32 of K) into LDS as Bs[n][k] ----
    {
      const int cg = (tid & 7) * 4;
#pragma unroll
      for (int p = 0; p < 4; ++p) {
        const int r = (tid >> 3) + p * 32;
        f32x4 d = *(const f32x4*)(W + (size_t)(bn0 + r) * K + k0 + cg);
        bf16x4 s;
        s[0] = (bf16)d[0]; s[1] = (bf16)d[1]; s[2] = (bf16)d[2]; s[3] = (bf16)d[3];
        *(bf16x4*)&Bs[r * LDA + cg] = s;
      }
    }
    if (k0 + BK < K) { // hint next tiles into cache (global_prefetch_b8)
      __builtin_prefetch(A + (size_t)(bm0 + (tid >> 3)) * K + k0 + BK, 0, 1);
      __builtin_prefetch(W + (size_t)(bn0 + (tid >> 3)) * K + k0 + BK, 0, 1);
    }
    __syncthreads();

    // ---- fragment loads + 8 WMMAs ----
    bf16x16 afr[4];
#pragma unroll
    for (int i = 0; i < 4; ++i)
      afr[i] = ld_fragA(&As[(wm * 64 + i * 16 + lane16) * LDA + halfsel * 8]);
    bf16x16 bfr[2];
#pragma unroll
    for (int j = 0; j < 2; ++j)
      bfr[j] = ld_fragB(&Bs[(wn * 32 + j * 16 + lane16) * LDA + halfsel * 8]);
#pragma unroll
    for (int i = 0; i < 4; ++i)
#pragma unroll
      for (int j = 0; j < 2; ++j)
        acc[i][j] = wmma_bf16(afr[i], bfr[j], acc[i][j]);
  }

  // ---- epilogue: C/D layout -> VGPR e, lanes0-15: M=e,N=lane; lanes16-31: M=8+e ----
#pragma unroll
  for (int j = 0; j < 2; ++j) {
    const int gn = bn0 + wn * 32 + j * 16 + lane16;
    const float bv = bias[gn];
#pragma unroll
    for (int i = 0; i < 4; ++i) {
      const int gmb = bm0 + wm * 64 + i * 16 + 8 * halfsel;
#pragma unroll
      for (int e = 0; e < 8; ++e)
        C[(size_t)(gmb + e) * N + gn] = (OT)(acc[i][j][e] + bv);
    }
  }
}

// ---------------------------------------------------------------------------
// Flash attention: block = (64 L-rows, one head, one batch), 128 threads.
// Q/K tiles staged by the Tensor Data Mover (tensor_load_to_lds + TENSORcnt);
// V staged with global_load_b128 + LDS transpose. All matmuls via
// v_wmma_f32_16x16x32_bf16 with online softmax.
// Output written in (b,h,l,e) flat order -> final GEMM reading it row-major
// reproduces the reference's transpose/view bug for free.
// ---------------------------------------------------------------------------
__global__ __launch_bounds__(128) void flash_attn_kernel(
    const bf16* __restrict__ Qp, const bf16* __restrict__ Kp,
    const bf16* __restrict__ Vp, bf16* __restrict__ O,
    int Bn, int Hn, int Ln, int Sn, int Dn)
{
  constexpr int E = 64, TS = 64, LDT = 72; // 72 bf16 = 144 B rows (16B aligned; TDM pad)
  __shared__ bf16 Qs[TS * LDT];
  __shared__ bf16 Ks[TS * LDT];
  __shared__ bf16 Vts[E * LDT];   // V transposed: [e][s]
  __shared__ bf16 Ps[TS * LDT];   // softmax probabilities, bf16

  const int tid     = threadIdx.x;
  const int wave    = tid >> 5;
  const int lane    = tid & 31;
  const int lane16  = lane & 15;
  const int halfsel = lane >> 4;
  const int l0 = blockIdx.x * TS;
  const int h  = blockIdx.y;
  const int b  = blockIdx.z;

  const size_t rowQ0 = (size_t)b * Ln;
  const size_t rowK0 = (size_t)b * Sn;
  const int colh = h * E;

  // ---- load Q tile (64 x 64 bf16) ----
#if USE_TDM
  if (wave == 0) {
    tdm_load_tile64x64((unsigned)(size_t)(void*)Qs,
                       Qp + (rowQ0 + l0) * Dn + colh, (unsigned)Ln);
    __builtin_amdgcn_s_wait_tensorcnt(0);
  }
#else
  {
    const int cg = (tid & 7) * 8;
#pragma unroll
    for (int p = 0; p < 4; ++p) {
      const int r = (tid >> 3) + p * 16;
      bf16x8 d = *(const bf16x8*)(Qp + (rowQ0 + l0 + r) * Dn + colh + cg);
      *(bf16x8*)&Qs[r * LDT + cg] = d;
    }
  }
#endif

  f32x8 o[4];
#pragma unroll
  for (int j = 0; j < 4; ++j) o[j] = {};
  float mrow[8], lrow[8];
#pragma unroll
  for (int i = 0; i < 8; ++i) { mrow[i] = -1e30f; lrow[i] = 0.f; }
  const float scale = 0.125f; // 1/sqrt(64)

  for (int s0 = 0; s0 < Sn; s0 += TS) {
    __syncthreads(); // previous iteration finished reading Ks/Vts
    // ---- K tile via Tensor Data Mover (one wave issues; TENSORcnt tracks) ----
#if USE_TDM
    if (wave == 0) {
      tdm_load_tile64x64((unsigned)(size_t)(void*)Ks,
                         Kp + (rowK0 + s0) * Dn + colh, (unsigned)Sn);
      __builtin_amdgcn_s_wait_tensorcnt(0);
    }
#endif
    // ---- V tile: global_load_b128, transposed into Vts[e][s] ----
    {
      const int cg = (tid & 7) * 8;
#pragma unroll
      for (int p = 0; p < 4; ++p) {
        const int r = (tid >> 3) + p * 16;
#if !USE_TDM
        bf16x8 dk = *(const bf16x8*)(Kp + (rowK0 + s0 + r) * Dn + colh + cg);
        *(bf16x8*)&Ks[r * LDT + cg] = dk;
#endif
        bf16x8 dv = *(const bf16x8*)(Vp + (rowK0 + s0 + r) * Dn + colh + cg);
#pragma unroll
        for (int i = 0; i < 8; ++i) Vts[(cg + i) * LDT + r] = dv[i];
      }
    }
    __syncthreads();

    // ---- scores = Q K^T : 16 rows x 64 s-cols per wave, K-dim = E = 2x32 ----
    f32x8 sc[4];
#pragma unroll
    for (int j = 0; j < 4; ++j) sc[j] = {};
#pragma unroll
    for (int ks = 0; ks < 2; ++ks) {
      bf16x16 aq = ld_fragA(&Qs[(wave * 16 + lane16) * LDT + ks * 32 + halfsel * 8]);
#pragma unroll
      for (int j = 0; j < 4; ++j) {
        bf16x16 bk = ld_fragB(&Ks[(j * 16 + lane16) * LDT + ks * 32 + halfsel * 8]);
        sc[j] = wmma_bf16(aq, bk, sc[j]);
      }
    }

    // ---- online softmax (row stats replicated across each 16-lane half) ----
#pragma unroll
    for (int i = 0; i < 8; ++i) {
      float pm = -1e30f;
#pragma unroll
      for (int j = 0; j < 4; ++j) pm = fmaxf(pm, sc[j][i]);
#pragma unroll
      for (int msk = 1; msk < 16; msk <<= 1)
        pm = fmaxf(pm, __shfl_xor(pm, msk, 32));
      const float mnew  = fmaxf(mrow[i], pm * scale);
      const float alpha = __expf(mrow[i] - mnew);
      const int prow = wave * 16 + i + 8 * halfsel;
      float psum = 0.f;
#pragma unroll
      for (int j = 0; j < 4; ++j) {
        const float pv = __expf(sc[j][i] * scale - mnew);
        psum += pv;
        Ps[prow * LDT + j * 16 + lane16] = (bf16)pv;
      }
#pragma unroll
      for (int msk = 1; msk < 16; msk <<= 1)
        psum += __shfl_xor(psum, msk, 32);
      lrow[i] = lrow[i] * alpha + psum;
      mrow[i] = mnew;
#pragma unroll
      for (int j = 0; j < 4; ++j) o[j][i] *= alpha;
    }

    // ---- O += P @ V  (P rows are wave-private; same-wave LDS ops are in-order) ----
#pragma unroll
    for (int ks = 0; ks < 2; ++ks) {
      bf16x16 ap = ld_fragA(&Ps[(wave * 16 + lane16) * LDT + ks * 32 + halfsel * 8]);
#pragma unroll
      for (int j = 0; j < 4; ++j) {
        bf16x16 bv = ld_fragB(&Vts[(j * 16 + lane16) * LDT + ks * 32 + halfsel * 8]);
        o[j] = wmma_bf16(ap, bv, o[j]);
      }
    }
  }

  // ---- normalize and store in (b,h,l,e) flat order ----
#pragma unroll
  for (int i = 0; i < 8; ++i) {
    const int gl = l0 + wave * 16 + i + 8 * halfsel;
    const float inv = 1.f / lrow[i];
    const size_t base = (((size_t)b * Hn + h) * Ln + gl) * E;
#pragma unroll
    for (int j = 0; j < 4; ++j)
      O[base + j * 16 + lane16] = (bf16)(o[j][i] * inv);
  }
}

// ---------------------------------------------------------------------------
extern "C" void kernel_launch(void* const* d_in, const int* in_sizes, int n_in,
                              void* d_out, int out_size, void* d_ws, size_t ws_size,
                              hipStream_t stream) {
  const float* q  = (const float*)d_in[0];
  const float* k  = (const float*)d_in[1];
  const float* v  = (const float*)d_in[2];
  const float* Wq = (const float*)d_in[3];
  const float* bq = (const float*)d_in[4];
  const float* Wo = (const float*)d_in[5];
  const float* bo = (const float*)d_in[6];
  float* out = (float*)d_out;

  const int B = 4, L = 2048, S = 2048, D = 1024, H = 16;
  const int M = B * L; // 8192 rows for all projection GEMMs

  bf16* qp = (bf16*)d_ws;                  // (B,L,D)  bf16
  bf16* kp = qp + (size_t)M * D;           // (B,S,D)  bf16
  bf16* vp = kp + (size_t)M * D;           // (B,S,D)  bf16
  bf16* ao = vp + (size_t)M * D;           // (B,H,L,E) flat bf16 (= scrambled (M,D))

  dim3 ggrid(M / 128, D / 128);
  // Projections: q/k/v all through Wq (faithful to the reference's bug).
  gemm_bt_bias<float, bf16><<<ggrid, 256, 0, stream>>>(q, Wq, bq, qp, M, D, D);
  gemm_bt_bias<float, bf16><<<ggrid, 256, 0, stream>>>(k, Wq, bq, kp, M, D, D);
  gemm_bt_bias<float, bf16><<<ggrid, 256, 0, stream>>>(v, Wq, bq, vp, M, D, D);
  // Fused attention (never materializes the 64 GB score tensor).
  flash_attn_kernel<<<dim3(L / 64, H, B), 128, 0, stream>>>(qp, kp, vp, ao, B, H, L, S, D);
  // Output projection reads ao row-major -> implements the transpose/view bug.
  gemm_bt_bias<bf16, float><<<ggrid, 256, 0, stream>>>(ao, Wo, bo, out, M, D, D);
}
